// Net_29025388986625
// MI455X (gfx1250) — matrix operand order repaired and checked by reading, not compiled, not run on hardware
//
#include <hip/hip_runtime.h>

#define H       128
#define GATES   512
#define KPAD    160
#define NFEATS  16
#define EFEATS  20
#define NCLS    4
#define NITER   20
#define NEG_SLOPE 0.01f

typedef __bf16 bf16;
typedef __attribute__((ext_vector_type(16))) __bf16 v16bf;
typedef __attribute__((ext_vector_type(8)))  float  v8f;

__device__ __forceinline__ int imin(int a, int b) { return a < b ? a : b; }
__device__ __forceinline__ int imax(int a, int b) { return a > b ? a : b; }

__device__ __forceinline__ v8f vzero8() {
    v8f z;
#pragma unroll
    for (int i = 0; i < 8; ++i) z[i] = 0.0f;
    return z;
}

// A fragment (16x32 bf16): lane m=l&15, kb=(l>>4)*16; 16 consecutive K per lane
__device__ __forceinline__ v16bf load_a_frag(const bf16* p) {
    union { v16bf v; uint4 q[2]; } u;
    u.q[0] = *(const uint4*)(p);
    u.q[1] = *(const uint4*)(p + 8);
    return u.v;
}
// B fragment (32x16 bf16): lane n=l&15, kb2=(l>>4)*16; 16 consecutive K per lane
__device__ __forceinline__ v16bf load_b_frag(const bf16* p) {
    union { v16bf v; uint4 q[2]; } u;
    u.q[0] = *(const uint4*)(p);
    u.q[1] = *(const uint4*)(p + 8);
    return u.v;
}

// 3-product bf16 split GEMM step: acc += Ahi*Bhi + Ahi*Blo + Alo*Bhi
__device__ __forceinline__ v8f wmma3(v16bf ah, v16bf al, v16bf bh, v16bf bl, v8f acc) {
    acc = __builtin_amdgcn_wmma_f32_16x16x32_bf16(false, ah, false, bh, (short)0, acc, false, false);
    acc = __builtin_amdgcn_wmma_f32_16x16x32_bf16(false, ah, false, bl, (short)0, acc, false, false);
    acc = __builtin_amdgcn_wmma_f32_16x16x32_bf16(false, al, false, bh, (short)0, acc, false, false);
    return acc;
}

__device__ __forceinline__ void bsplit(float v, bf16& hi, bf16& lo) {
    hi = (bf16)v;
    lo = (bf16)(v - (float)hi);
}
__device__ __forceinline__ float sigm(float x) { return 1.0f / (1.0f + __expf(-x)); }
__device__ __forceinline__ float lrelu(float x) { return x >= 0.0f ? x : NEG_SLOPE * x; }

// ---------------------------------------------------------------- utility
__global__ void zero32(unsigned int* p, long n) {
    long i = (long)blockIdx.x * blockDim.x + threadIdx.x;
    long s = (long)gridDim.x * blockDim.x;
    for (; i < n; i += s) p[i] = 0u;
}

__global__ void count_kernel(int E_, const int* __restrict__ dst, int* counts) {
    int i = blockIdx.x * blockDim.x + threadIdx.x;
    if (i < E_) atomicAdd(&counts[dst[i]], 1);
}

__global__ void fill_node_feats(int N_, const float* __restrict__ x, bf16* hi, bf16* lo) {
    int i = blockIdx.x * blockDim.x + threadIdx.x;
    if (i >= N_ * NFEATS) return;
    int n = i / NFEATS, j = i % NFEATS;
    bf16 h, l; bsplit(x[i], h, l);
    hi[(size_t)n * KPAD + j] = h;
    lo[(size_t)n * KPAD + j] = l;
}

// edge_feat = edge_attr @ W_fc1.T + b_fc1
__global__ void fill_edge_feats(int E_, const float* __restrict__ ea,
                                const float* __restrict__ Wfc1, const float* __restrict__ bfc1,
                                bf16* hi, bf16* lo) {
    int i = blockIdx.x * blockDim.x + threadIdx.x;
    if (i >= E_ * EFEATS) return;
    int e = i / EFEATS, j = i % EFEATS;
    float v = ea[(size_t)e * 2 + 0] * Wfc1[j * 2 + 0] + ea[(size_t)e * 2 + 1] * Wfc1[j * 2 + 1] + bfc1[j];
    bf16 h, l; bsplit(v, h, l);
    hi[(size_t)e * KPAD + j] = h;
    lo[(size_t)e * KPAD + j] = l;
}

// pack [GATES, KPAD]: k<in_w from W_ih, 32<=k<160 from W_hh, else 0
__global__ void pack_lstm_w(const float* __restrict__ Wih, const float* __restrict__ Whh,
                            int in_w, bf16* hi, bf16* lo) {
    int i = blockIdx.x * blockDim.x + threadIdx.x;
    if (i >= GATES * KPAD) return;
    int n = i / KPAD, k = i % KPAD;
    float v = 0.0f;
    if (k < in_w) v = Wih[n * in_w + k];
    else if (k >= 32) v = Whh[n * H + (k - 32)];
    bf16 h, l; bsplit(v, h, l);
    hi[i] = h; lo[i] = l;
}

__global__ void pack_plain_w(const float* __restrict__ W, int rows, int realRows, int K,
                             bf16* hi, bf16* lo) {
    int i = blockIdx.x * blockDim.x + threadIdx.x;
    if (i >= rows * K) return;
    int n = i / K, k = i % K;
    float v = (n < realRows) ? W[n * K + k] : 0.0f;
    bf16 h, l; bsplit(v, h, l);
    hi[i] = h; lo[i] = l;
}

__global__ void bias_sum(const float* a, const float* b, float* o) {
    int i = blockIdx.x * blockDim.x + threadIdx.x;
    if (i < GATES) o[i] = a[i] + b[i];
}

// ---------------------------------------------------------------- LSTM cell
// One block = 16 rows; 8 waves x 64 gate columns. Gates staged in LDS, then
// elementwise i/f/g/o update writes c (fp32) and h (fp32 opt + bf16 hi/lo).
__global__ void __launch_bounds__(256) lstm_kernel(
    int rows,
    const bf16* __restrict__ Ahi, const bf16* __restrict__ Alo,   // [rows,KPAD]
    const bf16* __restrict__ Whi, const bf16* __restrict__ Wlo,   // [GATES,KPAD]
    const float* __restrict__ bsum,                               // [GATES]
    float* __restrict__ c,                                        // [rows,H]
    float* __restrict__ hf32,                                     // [rows,H] or null
    bf16* h1hi, bf16* h1lo, int ld1,                              // dest1 (A array + 32)
    bf16* h2hi, bf16* h2lo, int ld2)                              // dest2 or null
{
    __shared__ float gsm[16 * GATES];
    int lane = threadIdx.x & 31;
    int wv   = threadIdx.x >> 5;
    int r0   = blockIdx.x * 16;
    int c0   = wv * 64;

    int m   = lane & 15;
    int row = imin(r0 + m, rows - 1);
    int kb  = (lane >> 4) * 16;
    int nB  = lane & 15;
    int kb2 = (lane >> 4) * 16;

    v8f acc[4];
#pragma unroll
    for (int t = 0; t < 4; ++t) acc[t] = vzero8();

    for (int kt = 0; kt < KPAD / 32; ++kt) {
        int k0 = kt * 32;
        const bf16* ap = Ahi + (size_t)row * KPAD + k0 + kb;
        const bf16* lp = Alo + (size_t)row * KPAD + k0 + kb;
        v16bf afh = load_a_frag(ap);
        v16bf afl = load_a_frag(lp);
#pragma unroll
        for (int nt = 0; nt < 4; ++nt) {
            int ncol = c0 + nt * 16 + nB;
            const bf16* bp = Whi + (size_t)ncol * KPAD + k0 + kb2;
            const bf16* bq = Wlo + (size_t)ncol * KPAD + k0 + kb2;
            v16bf bfh = load_b_frag(bp);
            v16bf bfl = load_b_frag(bq);
            acc[nt] = wmma3(afh, afl, bfh, bfl, acc[nt]);
        }
    }

    int mb = (lane >> 4) * 8;
#pragma unroll
    for (int nt = 0; nt < 4; ++nt) {
        int col = c0 + nt * 16 + nB;
#pragma unroll
        for (int i = 0; i < 8; ++i)
            gsm[(mb + i) * GATES + col] = acc[nt][i];
    }
    __syncthreads();

    for (int idx = threadIdx.x; idx < 16 * H; idx += 256) {
        int r = idx >> 7, h = idx & (H - 1);
        int rr = r0 + r;
        if (rr >= rows) continue;
        float ig = gsm[r * GATES + h]           + bsum[h];
        float fg = gsm[r * GATES + H + h]       + bsum[H + h];
        float gg = gsm[r * GATES + 2 * H + h]   + bsum[2 * H + h];
        float og = gsm[r * GATES + 3 * H + h]   + bsum[3 * H + h];
        size_t ci = (size_t)rr * H + h;
        float cn = sigm(fg) * c[ci] + sigm(ig) * tanhf(gg);
        float hn = sigm(og) * tanhf(cn);
        c[ci] = cn;
        if (hf32) hf32[ci] = hn;
        bf16 hh, hl; bsplit(hn, hh, hl);
        h1hi[(size_t)rr * ld1 + h] = hh;
        h1lo[(size_t)rr * ld1 + h] = hl;
        if (h2hi) {
            h2hi[(size_t)rr * ld2 + h] = hh;
            h2lo[(size_t)rr * ld2 + h] = hl;
        }
    }
}

// ---------------------------------------------------------------- edge message
// edge_msg = lrelu([h_n[src] | h_e | h_n[dst]] @ W_edge_mpn.T + b); writes next edge input.
__global__ void __launch_bounds__(256) edge_msg_kernel(
    int E_, const int* __restrict__ src, const int* __restrict__ dst,
    const bf16* __restrict__ nHhi, const bf16* __restrict__ nHlo,   // nodeA+32, ld KPAD
    const bf16* __restrict__ eHhi, const bf16* __restrict__ eHlo,   // edgeA+32, ld KPAD
    const bf16* __restrict__ Whi, const bf16* __restrict__ Wlo,     // [32,384]
    const float* __restrict__ bias,                                 // [EFEATS]
    bf16* outHi, bf16* outLo)                                       // edgeA base, ld KPAD
{
    int lane = threadIdx.x & 31;
    int wv   = threadIdx.x >> 5;
    int tile = blockIdx.x * 8 + wv;
    int e0   = tile * 16;
    if (e0 >= E_) return;

    int m = lane & 15;
    int e = imin(e0 + m, E_ - 1);
    int s = src[e], d = dst[e];
    int kb  = (lane >> 4) * 16;
    int nB  = lane & 15;
    int kb2 = (lane >> 4) * 16;

    const bf16* bHi[3] = { nHhi + (size_t)s * KPAD, eHhi + (size_t)e * KPAD, nHhi + (size_t)d * KPAD };
    const bf16* bLo[3] = { nHlo + (size_t)s * KPAD, eHlo + (size_t)e * KPAD, nHlo + (size_t)d * KPAD };

    v8f acc0 = vzero8(), acc1 = vzero8();
    for (int kt = 0; kt < 12; ++kt) {
        int reg = kt >> 2;
        int k0  = (kt & 3) * 32;
        v16bf afh = load_a_frag(bHi[reg] + k0 + kb);
        v16bf afl = load_a_frag(bLo[reg] + k0 + kb);
        int kg = kt * 32;
        {
            v16bf bfh = load_b_frag(Whi + (size_t)nB * 384 + kg + kb2);
            v16bf bfl = load_b_frag(Wlo + (size_t)nB * 384 + kg + kb2);
            acc0 = wmma3(afh, afl, bfh, bfl, acc0);
        }
        {
            v16bf bfh = load_b_frag(Whi + (size_t)(16 + nB) * 384 + kg + kb2);
            v16bf bfl = load_b_frag(Wlo + (size_t)(16 + nB) * 384 + kg + kb2);
            acc1 = wmma3(afh, afl, bfh, bfl, acc1);
        }
    }

    int mb = (lane >> 4) * 8;
#pragma unroll
    for (int i = 0; i < 8; ++i) {
        int er = e0 + mb + i;
        if (er >= E_) continue;
        float v0 = lrelu(acc0[i] + bias[nB]);
        bf16 h, l; bsplit(v0, h, l);
        outHi[(size_t)er * KPAD + nB] = h;
        outLo[(size_t)er * KPAD + nB] = l;
        int col1 = 16 + nB;
        if (col1 < EFEATS) {
            float v1 = lrelu(acc1[i] + bias[col1]);
            bsplit(v1, h, l);
            outHi[(size_t)er * KPAD + col1] = h;
            outLo[(size_t)er * KPAD + col1] = l;
        }
    }
}

// ---------------------------------------------------------------- node message
// node_msg = lrelu([agg | h_n] @ W_node_mpn.T + b); writes next node input.
__global__ void __launch_bounds__(256) node_msg_kernel(
    int N_,
    const bf16* __restrict__ Ahi, const bf16* __restrict__ Alo,   // [N,256]
    const bf16* __restrict__ Whi, const bf16* __restrict__ Wlo,   // [16,256]
    const float* __restrict__ bias,                               // [NFEATS]
    bf16* outHi, bf16* outLo)                                     // nodeA base, ld KPAD
{
    int lane = threadIdx.x & 31;
    int wv   = threadIdx.x >> 5;
    int tile = blockIdx.x * 8 + wv;
    int n0   = tile * 16;
    if (n0 >= N_) return;

    int m   = lane & 15;
    int row = imin(n0 + m, N_ - 1);
    int kb  = (lane >> 4) * 16;
    int nB  = lane & 15;
    int kb2 = (lane >> 4) * 16;

    v8f acc = vzero8();
    for (int kt = 0; kt < 8; ++kt) {
        int k0 = kt * 32;
        v16bf afh = load_a_frag(Ahi + (size_t)row * 256 + k0 + kb);
        v16bf afl = load_a_frag(Alo + (size_t)row * 256 + k0 + kb);
        v16bf bfh = load_b_frag(Whi + (size_t)nB * 256 + k0 + kb2);
        v16bf bfl = load_b_frag(Wlo + (size_t)nB * 256 + k0 + kb2);
        acc = wmma3(afh, afl, bfh, bfl, acc);
    }

    int mb = (lane >> 4) * 8;
#pragma unroll
    for (int i = 0; i < 8; ++i) {
        int nr = n0 + mb + i;
        if (nr >= N_) continue;
        float v = lrelu(acc[i] + bias[nB]);
        bf16 h, l; bsplit(v, h, l);
        outHi[(size_t)nr * KPAD + nB] = h;
        outLo[(size_t)nr * KPAD + nB] = l;
    }
}

// ---------------------------------------------------------------- scatter-mean
__global__ void scatter_kernel(int E_, const int* __restrict__ dst,
                               const bf16* __restrict__ hhi, const bf16* __restrict__ hlo,
                               float* agg) {
    int tid = blockIdx.x * blockDim.x + threadIdx.x;
    if (tid >= E_ * H) return;
    int e = tid >> 7, h = tid & (H - 1);
    float v = (float)hhi[(size_t)e * KPAD + h] + (float)hlo[(size_t)e * KPAD + h];
    atomicAdd(&agg[(size_t)dst[e] * H + h], v);
}

__global__ void finalize_kernel(int N_, const float* __restrict__ agg,
                                const int* __restrict__ counts,
                                bf16* outHi, bf16* outLo) {     // nmsgA base, ld 256
    int tid = blockIdx.x * blockDim.x + threadIdx.x;
    if (tid >= N_ * H) return;
    int n = tid >> 7, h = tid & (H - 1);
    float v = agg[tid] / (float)imax(counts[n], 1);
    bf16 hh, hl; bsplit(v, hh, hl);
    outHi[(size_t)n * 256 + h] = hh;
    outLo[(size_t)n * 256 + h] = hl;
}

// ---------------------------------------------------------------- prediction head
__global__ void pred_kernel(int N_, const float* __restrict__ hN,
                            const float* __restrict__ Wp, const float* __restrict__ bp,
                            float* out) {
    int tid = blockIdx.x * blockDim.x + threadIdx.x;
    if (tid >= N_ * NCLS) return;
    int n = tid >> 2, cls = tid & 3;
    const float* hr = hN + (size_t)n * H;
    const float* wr = Wp + cls * H;
    float s = bp[cls];
#pragma unroll 8
    for (int k = 0; k < H; ++k) s += hr[k] * wr[k];
    out[tid] += s;
}

// ---------------------------------------------------------------- launch
extern "C" void kernel_launch(void* const* d_in, const int* in_sizes, int n_in,
                              void* d_out, int out_size, void* d_ws, size_t ws_size,
                              hipStream_t stream) {
    (void)n_in; (void)out_size; (void)ws_size;
    const float* x     = (const float*)d_in[0];
    const int*   eidx  = (const int*)d_in[1];
    const float* eattr = (const float*)d_in[2];
    const float* Wihn  = (const float*)d_in[3];
    const float* Whhn  = (const float*)d_in[4];
    const float* bihn  = (const float*)d_in[5];
    const float* bhhn  = (const float*)d_in[6];
    const float* Wihe  = (const float*)d_in[7];
    const float* Whhe  = (const float*)d_in[8];
    const float* bihe  = (const float*)d_in[9];
    const float* bhhe  = (const float*)d_in[10];
    const float* Wnm   = (const float*)d_in[11];
    const float* bnm   = (const float*)d_in[12];
    const float* Wem   = (const float*)d_in[13];
    const float* bem   = (const float*)d_in[14];
    const float* Wfc1  = (const float*)d_in[15];
    const float* bfc1  = (const float*)d_in[16];
    const float* Wpred = (const float*)d_in[17];
    const float* bpred = (const float*)d_in[18];

    int N = in_sizes[0] / NFEATS;
    int E = in_sizes[1] / 2;
    const int* src = eidx;
    const int* dst = eidx + E;
    float* out = (float*)d_out;

    char* base = (char*)d_ws;
    size_t off = 0;
    auto alloc = [&](size_t bytes) -> char* {
        char* r = base + off;
        off += (bytes + 255) & ~(size_t)255;
        return r;
    };

    bf16* eAhi = (bf16*)alloc((size_t)E * KPAD * sizeof(bf16));
    bf16* eAlo = (bf16*)alloc((size_t)E * KPAD * sizeof(bf16));
    bf16* nAhi = (bf16*)alloc((size_t)N * KPAD * sizeof(bf16));
    bf16* nAlo = (bf16*)alloc((size_t)N * KPAD * sizeof(bf16));
    bf16* mAhi = (bf16*)alloc((size_t)N * 256 * sizeof(bf16));
    bf16* mAlo = (bf16*)alloc((size_t)N * 256 * sizeof(bf16));
    float* cE  = (float*)alloc((size_t)E * H * sizeof(float));
    float* cN  = (float*)alloc((size_t)N * H * sizeof(float));
    float* hN  = (float*)alloc((size_t)N * H * sizeof(float));
    float* agg = (float*)alloc((size_t)N * H * sizeof(float));
    int*  counts = (int*)alloc((size_t)N * sizeof(int));
    bf16* WnHi = (bf16*)alloc((size_t)GATES * KPAD * 2);
    bf16* WnLo = (bf16*)alloc((size_t)GATES * KPAD * 2);
    bf16* WeHi = (bf16*)alloc((size_t)GATES * KPAD * 2);
    bf16* WeLo = (bf16*)alloc((size_t)GATES * KPAD * 2);
    bf16* WmHi = (bf16*)alloc((size_t)32 * 384 * 2);
    bf16* WmLo = (bf16*)alloc((size_t)32 * 384 * 2);
    bf16* WqHi = (bf16*)alloc((size_t)16 * 256 * 2);
    bf16* WqLo = (bf16*)alloc((size_t)16 * 256 * 2);
    float* bsN = (float*)alloc(GATES * sizeof(float));
    float* bsE = (float*)alloc(GATES * sizeof(float));

    // zero entire used scratch region + output accumulator
    zero32<<<2048, 256, 0, stream>>>((unsigned int*)d_ws, (long)(off / 4));
    zero32<<<(N * NCLS + 255) / 256, 256, 0, stream>>>((unsigned int*)out, (long)(N * NCLS));

    // initial features + weight packing + counts
    fill_node_feats<<<(N * NFEATS + 255) / 256, 256, 0, stream>>>(N, x, nAhi, nAlo);
    fill_edge_feats<<<(E * EFEATS + 255) / 256, 256, 0, stream>>>(E, eattr, Wfc1, bfc1, eAhi, eAlo);
    pack_lstm_w<<<(GATES * KPAD + 255) / 256, 256, 0, stream>>>(Wihn, Whhn, NFEATS, WnHi, WnLo);
    pack_lstm_w<<<(GATES * KPAD + 255) / 256, 256, 0, stream>>>(Wihe, Whhe, EFEATS, WeHi, WeLo);
    pack_plain_w<<<(32 * 384 + 255) / 256, 256, 0, stream>>>(Wem, 32, EFEATS, 384, WmHi, WmLo);
    pack_plain_w<<<(16 * 256 + 255) / 256, 256, 0, stream>>>(Wnm, 16, NFEATS, 256, WqHi, WqLo);
    bias_sum<<<2, 256, 0, stream>>>(bihn, bhhn, bsN);
    bias_sum<<<2, 256, 0, stream>>>(bihe, bhhe, bsE);
    count_kernel<<<(E + 255) / 256, 256, 0, stream>>>(E, dst, counts);

    int nTiles = (N + 15) / 16;
    int eTiles = (E + 15) / 16;

    for (int t = 0; t < NITER; ++t) {
        // node LSTM: h_n -> nodeA(+32), nmsgA(+128), fp32 hN
        lstm_kernel<<<nTiles, 256, 0, stream>>>(
            N, nAhi, nAlo, WnHi, WnLo, bsN, cN, hN,
            nAhi + 32, nAlo + 32, KPAD, mAhi + H, mAlo + H, 256);
        // edge LSTM: h_e -> edgeA(+32)
        lstm_kernel<<<eTiles, 256, 0, stream>>>(
            E, eAhi, eAlo, WeHi, WeLo, bsE, cE, (float*)nullptr,
            eAhi + 32, eAlo + 32, KPAD, (bf16*)nullptr, (bf16*)nullptr, 0);
        // edge message -> next edge LSTM input
        edge_msg_kernel<<<(eTiles + 7) / 8, 256, 0, stream>>>(
            E, src, dst, nAhi + 32, nAlo + 32, eAhi + 32, eAlo + 32,
            WmHi, WmLo, bem, eAhi, eAlo);
        // scatter-mean of h_e into nodes
        zero32<<<(N * H + 255) / 256, 256, 0, stream>>>((unsigned int*)agg, (long)N * H);
        scatter_kernel<<<(E * H + 255) / 256, 256, 0, stream>>>(E, dst, eAhi + 32, eAlo + 32, agg);
        finalize_kernel<<<(N * H + 255) / 256, 256, 0, stream>>>(N, agg, counts, mAhi, mAlo);
        // node message -> next node LSTM input
        node_msg_kernel<<<(nTiles + 7) / 8, 256, 0, stream>>>(
            N, mAhi, mAlo, WqHi, WqLo, bnm, nAhi, nAlo);
        // per-iteration prediction head
        pred_kernel<<<(N * NCLS + 255) / 256, 256, 0, stream>>>(
            N, hN, Wpred + (size_t)t * NCLS * H, bpred + (size_t)t * NCLS, out);
    }
}
